// AqGeometricProcessor_69320772158261
// MI455X (gfx1250) — matrix operand order repaired and checked by reading, not compiled, not run on hardware
//
#include <hip/hip_runtime.h>
#include <hip/hip_bf16.h>

#define D 128
#define TILE 16
#define APAD_STRIDE 136   // 128 + 8 shorts pad -> 68 dwords row stride: conflict-free b128 DS reads

typedef __attribute__((ext_vector_type(16))) __bf16 v16bf;
typedef __attribute__((ext_vector_type(8)))  __bf16 v8bf;
typedef __attribute__((ext_vector_type(8)))  float  v8f;

__device__ __forceinline__ unsigned short f32_to_bf16(float f) {
    unsigned int u = __float_as_uint(f);
    u += 0x7FFFu + ((u >> 16) & 1u);   // round-to-nearest-even
    return (unsigned short)(u >> 16);
}

// ---------------- degree / norm precompute ----------------
__global__ void k_init_deg(float* deg, int n) {
    int i = blockIdx.x * blockDim.x + threadIdx.x;
    if (i < n) deg[i] = 1.0f;                     // self-loop contribution
}
__global__ void k_accum_deg(const int* __restrict__ dst, float* deg, int e) {
    int i = blockIdx.x * blockDim.x + threadIdx.x;
    if (i < e) atomicAdd(&deg[dst[i]], 1.0f);
}
__global__ void k_rsqrt_inplace(float* deg, int n) {
    int i = blockIdx.x * blockDim.x + threadIdx.x;
    if (i < n) deg[i] = rsqrtf(deg[i]);           // deg >= 1 always
}

// ---------------- fuse W = gcn_w @ lin_w, store bf16 row-major [j][k] ----------------
__global__ void k_fuse_w(const float* __restrict__ gcn_w, const float* __restrict__ lin_w,
                         unsigned short* __restrict__ wb, int total) {
    int t = blockIdx.x * blockDim.x + threadIdx.x;
    if (t >= total) return;
    int k = t & (D - 1);
    int j = (t >> 7) & (D - 1);
    int l = t >> 14;
    const float* g = gcn_w + (size_t)l * D * D + (size_t)j * D;
    const float* w = lin_w + (size_t)l * D * D + k;
    float s = 0.0f;
    #pragma unroll 8
    for (int m = 0; m < D; ++m) s += g[m] * w[(size_t)m * D];
    wb[t] = f32_to_bf16(s);
}

// ---------------- f32 -> bf16 conversion (vectorized x4) ----------------
__global__ void k_f32_to_bf16_4(const float4* __restrict__ in, ushort4* __restrict__ out, int n4) {
    int t = blockIdx.x * blockDim.x + threadIdx.x;
    if (t >= n4) return;
    float4 v = in[t];
    ushort4 o;
    o.x = f32_to_bf16(v.x); o.y = f32_to_bf16(v.y);
    o.z = f32_to_bf16(v.z); o.w = f32_to_bf16(v.w);
    out[t] = o;
}

// ---------------- WMMA GEMM: h[N x 128] = xb[N x 128] @ wb[128 x 128]^T (bf16 -> f32) ----------
// Block = 256 threads = 8 waves, covers 16 rows x 128 cols; wave w -> col tile w*16.
// A tile (16x128 bf16) is staged once into LDS (padded, bank-conflict-free) and
// shared by all 8 waves; B tile rows stream from global (each element used once per block).
__global__ void k_gemm_wmma(const unsigned short* __restrict__ xb,
                            const unsigned short* __restrict__ wb,
                            float* __restrict__ h, int n) {
    __shared__ unsigned short asmem[TILE * APAD_STRIDE];

    const int lane = threadIdx.x & 31;
    const int wave = threadIdx.x >> 5;
    const int row0 = blockIdx.x * TILE;
    const int col0 = wave * TILE;
    const int half = lane >> 4;     // 0: lanes 0-15, 1: lanes 16-31
    const int l15  = lane & 15;

    // --- cooperative A-tile stage: 256 threads x 16B chunks = 16 rows x 256B ---
    {
        const int t = threadIdx.x;
        const int r = t >> 4;              // 0..15
        const int c = (t & 15) * 8;        // short offset 0..120
        int grow = row0 + r;
        if (grow >= n) grow = n - 1;       // clamp loads (tail stores masked below)
        *(v8bf*)(&asmem[r * APAD_STRIDE + c]) = *(const v8bf*)(xb + (size_t)grow * D + c);
    }
    __syncthreads();

    const int col = col0 + l15;                       // always < 128
    const unsigned short* arow = asmem + l15 * APAD_STRIDE;
    const unsigned short* brow = wb + (size_t)col * D;

    v8f acc = {};
    #pragma unroll
    for (int k0 = 0; k0 < D; k0 += 32) {
        // A 16x32 bf16: lane holds K = ka..ka+7 and ka+16..ka+23, ka = k0 + half*8
        const int ka = k0 + (half ? 8 : 0);
        v8bf a0 = *(const v8bf*)(arow + ka);
        v8bf a1 = *(const v8bf*)(arow + ka + 16);
        v16bf A = __builtin_shufflevector(a0, a1, 0,1,2,3,4,5,6,7,8,9,10,11,12,13,14,15);
        // B 32x16 bf16: lane n holds K = kb..kb+7 and kb+16..kb+23, kb = k0 + half*8
        const int kb = k0 + (half ? 8 : 0);
        v8bf b0 = *(const v8bf*)(brow + kb);
        v8bf b1 = *(const v8bf*)(brow + kb + 16);
        v16bf B = __builtin_shufflevector(b0, b1, 0,1,2,3,4,5,6,7,8,9,10,11,12,13,14,15);
        acc = __builtin_amdgcn_wmma_f32_16x16x32_bf16(false, A, false, B, (short)0, acc,
                                                      false, false);
    }

    // D layout: VGPR r -> row row0 + half*8 + r, col = col0 + (lane&15)
    if (row0 + TILE <= n) {               // block-uniform fast path (no divergence)
        float* hrow = h + (size_t)(row0 + half * 8) * D + col;
        #pragma unroll
        for (int r = 0; r < 8; ++r) hrow[(size_t)r * D] = acc[r];
    } else {                              // tail block only
        #pragma unroll
        for (int r = 0; r < 8; ++r) {
            int rr = row0 + half * 8 + r;
            if (rr < n) h[(size_t)rr * D + col] = acc[r];
        }
    }
}

// ---------------- agg = bias + h * dinv^2 (self-loop fused with bias init) ----------------
__global__ void k_self_bias(const float* __restrict__ h, const float* __restrict__ dinv,
                            const float* __restrict__ bias, float* __restrict__ agg, int nd) {
    int t = blockIdx.x * blockDim.x + threadIdx.x;
    if (t >= nd) return;
    int i = t >> 7;
    int d = t & (D - 1);
    float di = dinv[i];
    agg[t] = bias[d] + h[t] * di * di;
}

// ---------------- edge scatter: agg[dst] += h[src] * dinv[src]*dinv[dst] ----------------
// 32 threads per edge, 4 floats per thread (coalesced float4 gather + 4 f32 atomics)
__global__ void k_scatter(const float* __restrict__ h, const float* __restrict__ dinv,
                          const int* __restrict__ src, const int* __restrict__ dst,
                          float* __restrict__ agg, int e) {
    int t = blockIdx.x * blockDim.x + threadIdx.x;
    int ee = t >> 5;
    if (ee >= e) return;
    int q = t & 31;
    int s = src[ee];
    int d = dst[ee];
    float c = dinv[s] * dinv[d];
    float4 hv = *((const float4*)(h + (size_t)s * D) + q);
    float* ap = agg + (size_t)d * D + (size_t)q * 4;
    atomicAdd(ap + 0, hv.x * c);
    atomicAdd(ap + 1, hv.y * c);
    atomicAdd(ap + 2, hv.z * c);
    atomicAdd(ap + 3, hv.w * c);
}

extern "C" void kernel_launch(void* const* d_in, const int* in_sizes, int n_in,
                              void* d_out, int out_size, void* d_ws, size_t ws_size,
                              hipStream_t stream) {
    const float* x     = (const float*)d_in[0];
    const int*   ei    = (const int*)  d_in[1];
    const float* lin_w = (const float*)d_in[2];
    const float* gcn_w = (const float*)d_in[3];
    const float* gcn_b = (const float*)d_in[4];

    const int N = in_sizes[0] / D;
    const int E = in_sizes[1] / 2;
    const int L = in_sizes[2] / (D * D);
    const int* src = ei;
    const int* dst = ei + E;

    // workspace carve-up (256B aligned)
    char* ws = (char*)d_ws;
    size_t off = 0;
    auto carve = [&](size_t bytes) {
        char* p = ws + off;
        off = (off + bytes + 255) & ~(size_t)255;
        return (void*)p;
    };
    float*          dinv = (float*)         carve((size_t)N * 4);
    unsigned short* wb   = (unsigned short*)carve((size_t)L * D * D * 2);
    unsigned short* xb   = (unsigned short*)carve((size_t)N * D * 2);
    float*          hbuf = (float*)         carve((size_t)N * D * 4);
    float*          bufA = (float*)         carve((size_t)N * D * 4);
    float*          bufB = (float*)         carve((size_t)N * D * 4);

    const int T = 256;
    const int nd = N * D;

    // 1) degree -> dinv (in place)
    k_init_deg     <<<(N + T - 1) / T, T, 0, stream>>>(dinv, N);
    k_accum_deg    <<<(E + T - 1) / T, T, 0, stream>>>(dst, dinv, E);
    k_rsqrt_inplace<<<(N + T - 1) / T, T, 0, stream>>>(dinv, N);

    // 2) fuse per-layer weights: W_l = gcn_w[l] @ lin_w[l]  (bf16, row-major [j][k])
    const int wtot = L * D * D;
    k_fuse_w<<<(wtot + T - 1) / T, T, 0, stream>>>(gcn_w, lin_w, wb, wtot);

    // 3) layers (ping-pong: x -> bufA -> bufB -> bufA -> d_out)
    for (int l = 0; l < L; ++l) {
        const float* xin = (l == 0) ? x : ((l & 1) ? bufA : bufB);
        float*       agg = (l == L - 1) ? (float*)d_out : ((l & 1) ? bufB : bufA);

        k_f32_to_bf16_4<<<(nd / 4 + T - 1) / T, T, 0, stream>>>(
            (const float4*)xin, (ushort4*)xb, nd / 4);

        k_gemm_wmma<<<(N + TILE - 1) / TILE, 256, 0, stream>>>(
            xb, wb + (size_t)l * D * D, hbuf, N);

        k_self_bias<<<(nd + T - 1) / T, T, 0, stream>>>(
            hbuf, dinv, gcn_b + (size_t)l * D, agg, nd);

        long long st = (long long)E * 32;
        k_scatter<<<(int)((st + T - 1) / T), T, 0, stream>>>(
            hbuf, dinv, src, dst, agg, E);
    }
}
